// Trans_with_A3D_30975304138813
// MI455X (gfx1250) — matrix operand order repaired and checked by reading, not compiled, not run on hardware
//
#include <hip/hip_runtime.h>

// ---------------- problem constants ----------------
constexpr int Bb    = 8;
constexpr int Ss    = 7;
constexpr int NTOK  = 1025;
constexpr int Cc    = 768;
constexpr int Hh    = 12;
constexpr int HD    = 64;
constexpr int MIDs  = 3;
constexpr int HID   = 3072;
constexpr int MROWS = Bb * NTOK;       // 8200
constexpr int NPAD  = 1056;            // 33*32, padded key count
constexpr float QK_SCALE = 0.125f;     // 64^-0.5

typedef __bf16 bf16;
typedef __attribute__((ext_vector_type(16))) __bf16 bfx16;
typedef __attribute__((ext_vector_type(8)))  __bf16 bfx8;
typedef __attribute__((ext_vector_type(8)))  float  v8f;

// ---- CDNA5 async global->LDS copy (ASYNCcnt-tracked, bypasses VGPRs) ----
// LDS aperture maps flat addr[31:0] to the LDS byte address (ISA 10.2), so the
// low 32 bits of a flat __shared__ pointer are a valid VDST LDS address.
__device__ __forceinline__ void async_copy_b128(unsigned lds_addr, const void* gaddr) {
  asm volatile("global_load_async_to_lds_b128 %0, %1, off"
               :: "v"(lds_addr), "v"(gaddr) : "memory");
}
__device__ __forceinline__ void wait_async0() {
  asm volatile("s_wait_asynccnt 0x0" ::: "memory");
}

// ---------------- fp32 -> bf16 convert ----------------
__global__ __launch_bounds__(256)
void f2bf_kernel(const float* __restrict__ in, bf16* __restrict__ out, int n) {
  int i = blockIdx.x * 256 + threadIdx.x;
  if (i < n) out[i] = (bf16)in[i];
}

// ---------------- LayerNorm 1: x (B,S,N,C) -> h_qk (B*N,6C) bf16, h_mid (B*N,C) bf16 ----
__global__ __launch_bounds__(256)
void ln1_kernel(const float* __restrict__ x, const float* __restrict__ gam,
                const float* __restrict__ bet, bf16* __restrict__ h_qk,
                bf16* __restrict__ h_mid) {
  __shared__ float red[256];
  const int row = blockIdx.x;                     // b*S*N + s*N + n
  const int b   = row / (Ss * NTOK);
  const int rem = row - b * Ss * NTOK;
  const int s   = rem / NTOK;
  const int n   = rem - s * NTOK;
  const int t   = threadIdx.x;
  const float* xr = x + (size_t)row * Cc;
  float l0 = xr[t], l1 = xr[t + 256], l2 = xr[t + 512];
  red[t] = l0 + l1 + l2;
  __syncthreads();
  for (int off = 128; off > 0; off >>= 1) {
    if (t < off) red[t] += red[t + off];
    __syncthreads();
  }
  const float mean = red[0] * (1.0f / Cc);
  __syncthreads();
  const float d0 = l0 - mean, d1 = l1 - mean, d2 = l2 - mean;
  red[t] = d0 * d0 + d1 * d1 + d2 * d2;
  __syncthreads();
  for (int off = 128; off > 0; off >>= 1) {
    if (t < off) red[t] += red[t + off];
    __syncthreads();
  }
  const float rstd = rsqrtf(red[0] * (1.0f / Cc) + 1e-6f);

  bf16* dst; size_t base;
  if (s == MIDs) { dst = h_mid; base = ((size_t)(b * NTOK + n)) * Cc; }
  else { int sp = (s < MIDs) ? s : s - 1;
         dst = h_qk; base = ((size_t)(b * NTOK + n)) * (6 * Cc) + (size_t)sp * Cc; }
  dst[base + t]       = (bf16)(d0 * rstd * gam[t]       + bet[t]);
  dst[base + t + 256] = (bf16)(d1 * rstd * gam[t + 256] + bet[t + 256]);
  dst[base + t + 512] = (bf16)(d2 * rstd * gam[t + 512] + bet[t + 512]);
}

// ---------------- LayerNorm 2: y (MROWS,C) f32 -> h2 (MROWS,C) bf16 ----------------
__global__ __launch_bounds__(256)
void ln2_kernel(const float* __restrict__ y, const float* __restrict__ gam,
                const float* __restrict__ bet, bf16* __restrict__ h2) {
  __shared__ float red[256];
  const int row = blockIdx.x;
  const int t = threadIdx.x;
  const float* yr = y + (size_t)row * Cc;
  float l0 = yr[t], l1 = yr[t + 256], l2 = yr[t + 512];
  red[t] = l0 + l1 + l2;
  __syncthreads();
  for (int off = 128; off > 0; off >>= 1) {
    if (t < off) red[t] += red[t + off];
    __syncthreads();
  }
  const float mean = red[0] * (1.0f / Cc);
  __syncthreads();
  const float d0 = l0 - mean, d1 = l1 - mean, d2 = l2 - mean;
  red[t] = d0 * d0 + d1 * d1 + d2 * d2;
  __syncthreads();
  for (int off = 128; off > 0; off >>= 1) {
    if (t < off) red[t] += red[t + off];
    __syncthreads();
  }
  const float rstd = rsqrtf(red[0] * (1.0f / Cc) + 1e-6f);
  bf16* dst = h2 + (size_t)row * Cc;
  dst[t]       = (bf16)(d0 * rstd * gam[t]       + bet[t]);
  dst[t + 256] = (bf16)(d1 * rstd * gam[t + 256] + bet[t + 256]);
  dst[t + 512] = (bf16)(d2 * rstd * gam[t + 512] + bet[t + 512]);
}

// ---------------- generic bf16 GEMM: C(M,N) = A(M,K) * B(N,K)^T, fp32 accum ----------
// 128x128 block tile, 8 waves (each 32x64 = 2x4 WMMA tiles), K-slabs of 32,
// double-buffered LDS filled by async global->LDS b128 copies.
enum { EP_QK = 0, EP_VT = 1, EP_PROJ = 2, EP_GELU = 3, EP_FC2 = 4 };

struct EpArgs {
  const float* bias;
  const float* resid;
  float* out_f;
  bf16*  out_b0;
  bf16*  out_b1;
};

constexpr int LDK = 40;   // padded LDS K-stride (bf16 elems)

template<int EPI>
__global__ __launch_bounds__(256)
void gemm_bf16_kernel(const bf16* __restrict__ A, const bf16* __restrict__ Bm,
                      int M, int N, int K, EpArgs ep) {
  __shared__ bf16 As[2][128 * LDK];
  __shared__ bf16 Bs[2][128 * LDK];
  const int tid  = threadIdx.x;
  const int lane = tid & 31;
  const int wave = tid >> 5;
  const int n0 = blockIdx.x * 128;
  const int m0 = blockIdx.y * 128;
  const int wr = wave & 3;    // m-offset wr*32 (2 sub-tiles)
  const int wc = wave >> 2;   // n-offset wc*64 (4 sub-tiles)

  v8f acc[2][4];
  #pragma unroll
  for (int i = 0; i < 2; i++)
    #pragma unroll
    for (int j = 0; j < 4; j++) acc[i][j] = {};

  // staging assignment: 2 threads per row, 16 bf16 (32B) each, for A and B
  const int ldrow = tid >> 1;
  const int ldcol = (tid & 1) * 16;
  int mg = m0 + ldrow; if (mg > M - 1) mg = M - 1;   // clamp: C rows >= M never stored
  const bf16* agp = A  + (size_t)mg * K + ldcol;
  const bf16* bgp = Bm + (size_t)(n0 + ldrow) * K + ldcol;
  unsigned aoff[2], boff[2];
  #pragma unroll
  for (int bidx = 0; bidx < 2; bidx++) {
    aoff[bidx] = (unsigned)(uintptr_t)&As[bidx][ldrow * LDK + ldcol];
    boff[bidx] = (unsigned)(uintptr_t)&Bs[bidx][ldrow * LDK + ldcol];
  }
  auto stage = [&](int bidx, int k0) {
    async_copy_b128(aoff[bidx],      agp + k0);
    async_copy_b128(aoff[bidx] + 16, agp + k0 + 8);
    async_copy_b128(boff[bidx],      bgp + k0);
    async_copy_b128(boff[bidx] + 16, bgp + k0 + 8);
  };

  const int mrow = lane & 15;
  const int kA0  = (lane < 16) ? 0 : 8;    // A layout: lo lanes K 0..7/16..23, hi 8..15/24..31
  const int kB   = (lane < 16) ? 0 : 8;    // B layout: lo lanes K 0..7/16..23, hi 8..15/24..31

  const int nk = K / 32;
  stage(0, 0);                              // prologue fill of buffer 0

  for (int kt = 0; kt < nk; kt++) {
    const int cur = kt & 1;
    wait_async0();        // my async copies into buf[cur] are complete
    __syncthreads();      // everyone's are; everyone done reading buf[cur^1]
    if (kt + 1 < nk) stage(cur ^ 1, (kt + 1) * 32);   // overlap next slab with compute

    bfx16 aF[2];
    #pragma unroll
    for (int ms = 0; ms < 2; ms++) {
      const bf16* ap = &As[cur][(wr * 32 + ms * 16 + mrow) * LDK];
      bfx8 c0 = *reinterpret_cast<const bfx8*>(ap + kA0);
      bfx8 c1 = *reinterpret_cast<const bfx8*>(ap + kA0 + 16);
      bfx16 a;
      #pragma unroll
      for (int i = 0; i < 8; i++) { a[i] = c0[i]; a[8 + i] = c1[i]; }
      aF[ms] = a;
    }
    bfx16 bF[4];
    #pragma unroll
    for (int ns = 0; ns < 4; ns++) {
      const bf16* bp = &Bs[cur][(wc * 64 + ns * 16 + mrow) * LDK + kB];
      bfx8 c0 = *reinterpret_cast<const bfx8*>(bp);
      bfx8 c1 = *reinterpret_cast<const bfx8*>(bp + 16);
      bfx16 b;
      #pragma unroll
      for (int i = 0; i < 8; i++) { b[i] = c0[i]; b[8 + i] = c1[i]; }
      bF[ns] = b;
    }
    #pragma unroll
    for (int ms = 0; ms < 2; ms++)
      #pragma unroll
      for (int ns = 0; ns < 4; ns++)
        acc[ms][ns] = __builtin_amdgcn_wmma_f32_16x16x32_bf16(
            false, aF[ms], false, bF[ns], (short)0, acc[ms][ns], false, false);
  }

  // ---- epilogue: C layout lane (row = r + (lane<16?0:8), col = lane&15) ----
  const int rowhi = (lane < 16) ? 0 : 8;
  const int ncl   = lane & 15;
  #pragma unroll
  for (int ms = 0; ms < 2; ms++) {
    #pragma unroll
    for (int ns = 0; ns < 4; ns++) {
      #pragma unroll
      for (int r = 0; r < 8; r++) {
        const int mgE = m0 + wr * 32 + ms * 16 + r + rowhi;
        const int ng  = n0 + wc * 64 + ns * 16 + ncl;
        if (mgE >= M) continue;
        const float v = acc[ms][ns][r];
        if constexpr (EPI == EP_QK) {
          const int b = mgE / NTOK, n = mgE - b * NTOK;
          const int tq = ng / Cc, jj = ng - tq * Cc;
          const int hh = jj >> 6, d = jj & 63;
          bf16* dst = tq ? ep.out_b1 : ep.out_b0;
          dst[(((size_t)(b * Hh + hh)) * NPAD + n) * HD + d] = (bf16)v;
        } else if constexpr (EPI == EP_VT) {
          const int b = mgE / NTOK, n = mgE - b * NTOK;
          const int hh = ng >> 6, d = ng & 63;
          ep.out_b0[(((size_t)(b * Hh + hh)) * HD + d) * NPAD + n] = (bf16)v;
        } else if constexpr (EPI == EP_PROJ) {
          const int b = mgE / NTOK, n = mgE - b * NTOK;
          const float res = ep.resid[(((size_t)(b * Ss + MIDs)) * NTOK + n) * Cc + ng];
          ep.out_f[(size_t)mgE * Cc + ng] = v + ep.bias[ng] + res;
        } else if constexpr (EPI == EP_GELU) {
          const float xg = v + ep.bias[ng];
          const float g = 0.5f * xg * (1.0f + erff(xg * 0.7071067811865475f));
          ep.out_b0[(size_t)mgE * HID + ng] = (bf16)g;
        } else { // EP_FC2
          ep.out_f[(size_t)mgE * Cc + ng] =
              v + ep.bias[ng] + ep.resid[(size_t)mgE * Cc + ng];
        }
      }
    }
  }
}

// ---------------- attention: per wave one 16-row q tile ----------------
__global__ __launch_bounds__(64)
void attn_kernel(const bf16* __restrict__ Q, const bf16* __restrict__ Km,
                 const bf16* __restrict__ VT, bf16* __restrict__ O) {
  extern __shared__ char smem[];
  const int lane = threadIdx.x & 31;
  const int wave = threadIdx.x >> 5;
  float* Srow = (float*)smem + (size_t)wave * 16 * NPAD;          // 16 x NPAD scores
  float* sums = (float*)(smem + 2 * 16 * NPAD * sizeof(float)) + wave * 16;
  const int bh = blockIdx.x;                                      // b*H + h
  const int m0 = blockIdx.y * 32 + wave * 16;
  if (m0 >= NTOK) return;

  const bf16* qb  = Q  + (size_t)bh * NPAD * HD;
  const bf16* kb  = Km + (size_t)bh * NPAD * HD;
  const bf16* vtb = VT + (size_t)bh * HD * NPAD;

  const int mrow = lane & 15;
  const int kA0  = (lane < 16) ? 0 : 8;
  const int kB   = (lane < 16) ? 0 : 8;
  const int ncl  = lane & 15;

  // q A-fragments for this row tile (K-dim = head dim, two 32-chunks)
  bfx16 qF[2];
  #pragma unroll
  for (int h2 = 0; h2 < 2; h2++) {
    const bf16* qp = qb + (size_t)(m0 + mrow) * HD + h2 * 32 + kA0;
    bfx8 c0 = *reinterpret_cast<const bfx8*>(qp);
    bfx8 c1 = *reinterpret_cast<const bfx8*>(qp + 16);
    bfx16 a;
    #pragma unroll
    for (int i = 0; i < 8; i++) { a[i] = c0[i]; a[8 + i] = c1[i]; }
    qF[h2] = a;
  }

  // ---- phase 1: scores S = q*k^T * scale into LDS ----
  for (int kt = 0; kt < NPAD / 16; kt++) {
    v8f c = {};
    #pragma unroll
    for (int h2 = 0; h2 < 2; h2++) {
      const bf16* kp = kb + (size_t)(kt * 16 + ncl) * HD + h2 * 32 + kB;
      bfx8 c0 = *reinterpret_cast<const bfx8*>(kp);
      bfx8 c1 = *reinterpret_cast<const bfx8*>(kp + 16);
      bfx16 bfr;
      #pragma unroll
      for (int i = 0; i < 8; i++) { bfr[i] = c0[i]; bfr[8 + i] = c1[i]; }
      c = __builtin_amdgcn_wmma_f32_16x16x32_bf16(false, qF[h2], false, bfr,
                                                  (short)0, c, false, false);
    }
    const int col = kt * 16 + ncl;
    const bool valid = (col < NTOK);
    #pragma unroll
    for (int r = 0; r < 8; r++) {
      const int m = r + ((lane < 16) ? 0 : 8);
      Srow[m * NPAD + col] = valid ? c[r] * QK_SCALE : -1e30f;
    }
  }
  asm volatile("s_wait_dscnt 0" ::: "memory");
  __builtin_amdgcn_wave_barrier();

  // ---- phase 2: softmax rows (unnormalized exp; 1/sum folded into store) ----
  if (lane < 16) {
    float* p = Srow + lane * NPAD;
    float mx = -1e30f;
    for (int k = 0; k < NPAD; k++) mx = fmaxf(mx, p[k]);
    float sm = 0.0f;
    for (int k = 0; k < NPAD; k++) { float e = __expf(p[k] - mx); p[k] = e; sm += e; }
    sums[lane] = sm;
  }
  asm volatile("s_wait_dscnt 0" ::: "memory");
  __builtin_amdgcn_wave_barrier();

  // ---- phase 3: O = P * V ----
  v8f acc[4];
  #pragma unroll
  for (int i = 0; i < 4; i++) acc[i] = {};
  for (int kt = 0; kt < NPAD / 32; kt++) {
    const int ks = kt * 32;
    const float* pp = Srow + mrow * NPAD + ks + kA0;
    bfx16 a;
    #pragma unroll
    for (int i = 0; i < 8; i++) { a[i] = (bf16)pp[i]; a[8 + i] = (bf16)pp[16 + i]; }
    #pragma unroll
    for (int ns = 0; ns < 4; ns++) {
      const bf16* vp = vtb + (size_t)(ns * 16 + ncl) * NPAD + ks + kB;
      bfx8 c0 = *reinterpret_cast<const bfx8*>(vp);
      bfx8 c1 = *reinterpret_cast<const bfx8*>(vp + 16);
      bfx16 bfr;
      #pragma unroll
      for (int i = 0; i < 8; i++) { bfr[i] = c0[i]; bfr[8 + i] = c1[i]; }
      acc[ns] = __builtin_amdgcn_wmma_f32_16x16x32_bf16(false, a, false, bfr,
                                                        (short)0, acc[ns], false, false);
    }
  }
  const int bb = bh / Hh, hh = bh - bb * Hh;
  #pragma unroll
  for (int ns = 0; ns < 4; ns++) {
    #pragma unroll
    for (int r = 0; r < 8; r++) {
      const int m = r + ((lane < 16) ? 0 : 8);
      const int mg = m0 + m;
      if (mg >= NTOK) continue;
      const float val = acc[ns][r] / sums[m];
      O[((size_t)(bb * NTOK + mg)) * Cc + hh * HD + ns * 16 + ncl] = (bf16)val;
    }
  }
}

// ---------------- host launcher ----------------
extern "C" void kernel_launch(void* const* d_in, const int* in_sizes, int n_in,
                              void* d_out, int out_size, void* d_ws, size_t ws_size,
                              hipStream_t stream) {
  (void)in_sizes; (void)n_in; (void)out_size; (void)ws_size;
  const float* x      = (const float*)d_in[0];
  const float* ln1_g  = (const float*)d_in[1];
  const float* ln1_b  = (const float*)d_in[2];
  const float* qk_w   = (const float*)d_in[3];
  const float* v_w    = (const float*)d_in[4];
  const float* proj_w = (const float*)d_in[5];
  const float* proj_b = (const float*)d_in[6];
  const float* ln2_g  = (const float*)d_in[7];
  const float* ln2_b  = (const float*)d_in[8];
  const float* fc1_w  = (const float*)d_in[9];
  const float* fc1_b  = (const float*)d_in[10];
  const float* fc2_w  = (const float*)d_in[11];
  const float* fc2_b  = (const float*)d_in[12];
  float* out = (float*)d_out;

  char* ws = (char*)d_ws;
  auto alloc = [&](size_t bytes) -> char* {
    char* p = ws; ws += (bytes + 255) & ~(size_t)255; return p;
  };
  bf16* h_qk   = (bf16*)alloc((size_t)MROWS * 6 * Cc * sizeof(bf16));
  bf16* h_mid  = (bf16*)alloc((size_t)MROWS * Cc * sizeof(bf16));
  bf16* qk_wb  = (bf16*)alloc((size_t)2 * Cc * 6 * Cc * sizeof(bf16));
  bf16* v_wb   = (bf16*)alloc((size_t)Cc * Cc * sizeof(bf16));
  bf16* proj_wb= (bf16*)alloc((size_t)Cc * Cc * sizeof(bf16));
  bf16* fc1_wb = (bf16*)alloc((size_t)HID * Cc * sizeof(bf16));
  bf16* fc2_wb = (bf16*)alloc((size_t)Cc * HID * sizeof(bf16));
  const size_t qkBytes = (size_t)Bb * Hh * NPAD * HD * sizeof(bf16);
  bf16* Qb  = (bf16*)alloc(qkBytes);
  bf16* Kb  = (bf16*)alloc(qkBytes);
  bf16* VTb = (bf16*)alloc(qkBytes);
  bf16* Ob  = (bf16*)alloc((size_t)MROWS * Cc * sizeof(bf16));
  float* yb = (float*)alloc((size_t)MROWS * Cc * sizeof(float));
  bf16* h2b = (bf16*)alloc((size_t)MROWS * Cc * sizeof(bf16));
  bf16* gb  = (bf16*)alloc((size_t)MROWS * HID * sizeof(bf16));

  // weight conversion
  auto conv = [&](const float* src, bf16* dst, int n) {
    f2bf_kernel<<<(n + 255) / 256, 256, 0, stream>>>(src, dst, n);
  };
  conv(qk_w,   qk_wb,  2 * Cc * 6 * Cc);
  conv(v_w,    v_wb,   Cc * Cc);
  conv(proj_w, proj_wb, Cc * Cc);
  conv(fc1_w,  fc1_wb, HID * Cc);
  conv(fc2_w,  fc2_wb, Cc * HID);

  // zero padded Q/K/VT (tail keys/rows)
  hipMemsetAsync(Qb, 0, qkBytes, stream);
  hipMemsetAsync(Kb, 0, qkBytes, stream);
  hipMemsetAsync(VTb, 0, qkBytes, stream);

  // LN1 -> h_qk, h_mid
  ln1_kernel<<<Bb * Ss * NTOK, 256, 0, stream>>>(x, ln1_g, ln1_b, h_qk, h_mid);

  const int mBlocks = (MROWS + 127) / 128;  // 65

  // qk GEMM -> Q, K
  { EpArgs ep{nullptr, nullptr, nullptr, Qb, Kb};
    gemm_bf16_kernel<EP_QK><<<dim3(2 * Cc / 128, mBlocks), 256, 0, stream>>>(
        h_qk, qk_wb, MROWS, 2 * Cc, 6 * Cc, ep); }

  // v GEMM -> VT
  { EpArgs ep{nullptr, nullptr, nullptr, VTb, nullptr};
    gemm_bf16_kernel<EP_VT><<<dim3(Cc / 128, mBlocks), 256, 0, stream>>>(
        h_mid, v_wb, MROWS, Cc, Cc, ep); }

  // attention -> O (bf16, (B,N,C))
  const int attnSmem = 2 * 16 * NPAD * (int)sizeof(float) + 2 * 16 * (int)sizeof(float);
  hipFuncSetAttribute((const void*)attn_kernel,
                      hipFuncAttributeMaxDynamicSharedMemorySize, attnSmem);
  attn_kernel<<<dim3(Bb * Hh, (NTOK + 31) / 32), 64, attnSmem, stream>>>(Qb, Kb, VTb, Ob);

  // proj GEMM -> y = O*proj_w^T + proj_b + mid_x
  { EpArgs ep{proj_b, x, yb, nullptr, nullptr};
    gemm_bf16_kernel<EP_PROJ><<<dim3(Cc / 128, mBlocks), 256, 0, stream>>>(
        Ob, proj_wb, MROWS, Cc, Cc, ep); }

  // LN2 -> h2
  ln2_kernel<<<MROWS, 256, 0, stream>>>(yb, ln2_g, ln2_b, h2b);

  // fc1 GEMM + GELU -> g
  { EpArgs ep{fc1_b, nullptr, nullptr, gb, nullptr};
    gemm_bf16_kernel<EP_GELU><<<dim3(HID / 128, mBlocks), 256, 0, stream>>>(
        h2b, fc1_wb, MROWS, HID, Cc, ep); }

  // fc2 GEMM -> out = g*fc2_w^T + fc2_b + y
  { EpArgs ep{fc2_b, yb, out, nullptr, nullptr};
    gemm_bf16_kernel<EP_FC2><<<dim3(Cc / 128, mBlocks), 256, 0, stream>>>(
        gb, fc2_wb, MROWS, Cc, HID, ep); }
}
